// MoELayer_29635274342853
// MI455X (gfx1250) — matrix operand order repaired and checked by reading, not compiled, not run on hardware
//
#include <hip/hip_runtime.h>

#define T_TOK 4096
#define D_DIM 1024
#define E_EXP 8
#define F_DIM 1706
#define FP_DIM 1728   // F padded to multiple of 64

typedef __bf16 bf16;
typedef __bf16 v16bf __attribute__((ext_vector_type(16)));
typedef __bf16 v8bf  __attribute__((ext_vector_type(8)));
typedef __bf16 v4bf  __attribute__((ext_vector_type(4)));
typedef float  v8f   __attribute__((ext_vector_type(8)));

// ---- WMMA fragment loaders (CDNA5 16x16x32 bf16 layouts) ----
// A (16x32, MxK): lane r (=lane&15) = row M; lanes<16: K {0-7,16-23}, lanes>=16: K {8-15,24-31}
__device__ __forceinline__ v16bf frag_a(const bf16* lds, int row, int stride, int half) {
    const bf16* p = lds + row * stride;
    v8bf lo = *(const v8bf*)(p + half * 8);
    v8bf hi = *(const v8bf*)(p + 16 + half * 8);
    return __builtin_shufflevector(lo, hi, 0,1,2,3,4,5,6,7,8,9,10,11,12,13,14,15);
}
// B (32x16, KxN) stored transposed in LDS as [N][K]: lane r = col N; lanes<16: K {0-7,16-23}, lanes>=16: K {8-15,24-31}
__device__ __forceinline__ v16bf frag_b(const bf16* lds, int col, int stride, int half) {
    const bf16* p = lds + col * stride + half * 8;
    v8bf lo = *(const v8bf*)(p);
    v8bf hi = *(const v8bf*)(p + 16);
    return __builtin_shufflevector(lo, hi, 0,1,2,3,4,5,6,7,8,9,10,11,12,13,14,15);
}

// Async global->LDS B128 copy (CDNA5 GLOBAL_LOAD_ASYNC_TO_LDS_B128, tracked by ASYNCcnt).
__device__ __forceinline__ void async_copy_b128(void* ldsPtr, const void* gPtr) {
    unsigned int ldsAddr = (unsigned int)(unsigned long long)(uintptr_t)ldsPtr;
    unsigned long long gAddr = (unsigned long long)(uintptr_t)gPtr;
    asm volatile("global_load_async_to_lds_b128 %0, %1, off"
                 :: "v"(ldsAddr), "v"(gAddr) : "memory");
}
__device__ __forceinline__ void async_wait0() {
    asm volatile("s_wait_asynccnt 0x0" ::: "memory");
}

// ---- Router: logits, top-2 softmax combine weights, z-loss / prob partial sums ----
__global__ __launch_bounds__(256) void moe_router(const float* __restrict__ x,
                                                  const float* __restrict__ Wr,
                                                  float* __restrict__ comb,
                                                  float* __restrict__ partials) {
    __shared__ float sWr[D_DIM * E_EXP];   // 32 KB
    __shared__ float red[256];
    int tid = threadIdx.x;
    int t = blockIdx.x * 256 + tid;
    for (int i = tid; i < D_DIM * E_EXP; i += 256) sWr[i] = Wr[i];
    __syncthreads();

    float lg[E_EXP];
#pragma unroll
    for (int e = 0; e < E_EXP; e++) lg[e] = 0.0f;
    const float* xp = x + (size_t)t * D_DIM;
    for (int d = 0; d < D_DIM; d += 4) {
        float4 xv = *(const float4*)(xp + d);
#pragma unroll
        for (int e = 0; e < E_EXP; e++) {
            lg[e] += xv.x * sWr[(d + 0) * E_EXP + e];
            lg[e] += xv.y * sWr[(d + 1) * E_EXP + e];
            lg[e] += xv.z * sWr[(d + 2) * E_EXP + e];
            lg[e] += xv.w * sWr[(d + 3) * E_EXP + e];
        }
    }
    float z = 0.0f;
#pragma unroll
    for (int e = 0; e < E_EXP; e++) z += lg[e] * lg[e];
    int i0 = 0; float v0 = lg[0];
#pragma unroll
    for (int e = 1; e < E_EXP; e++) if (lg[e] > v0) { v0 = lg[e]; i0 = e; }
    int i1 = -1; float v1 = -3.4e38f;
#pragma unroll
    for (int e = 0; e < E_EXP; e++) if (e != i0 && lg[e] > v1) { v1 = lg[e]; i1 = e; }
    float w0 = 1.0f / (1.0f + __expf(v1 - v0));
    float w1 = 1.0f - w0;
    float p[E_EXP], s = 0.0f;
#pragma unroll
    for (int e = 0; e < E_EXP; e++) { p[e] = __expf(lg[e] - v0); s += p[e]; }
    float inv = 1.0f / s;
#pragma unroll
    for (int e = 0; e < E_EXP; e++) {
        p[e] *= inv;
        comb[(size_t)t * E_EXP + e] = (e == i0) ? w0 : ((e == i1) ? w1 : 0.0f);
    }
    float vals[9];
    vals[0] = z;
#pragma unroll
    for (int e = 0; e < E_EXP; e++) vals[1 + e] = p[e];
    for (int j = 0; j < 9; j++) {
        red[tid] = vals[j];
        __syncthreads();
        for (int sft = 128; sft > 0; sft >>= 1) {
            if (tid < sft) red[tid] += red[tid + sft];
            __syncthreads();
        }
        if (tid == 0) partials[blockIdx.x * 9 + j] = red[0];
        __syncthreads();
    }
}

// ---- x -> bf16 ----
__global__ __launch_bounds__(256) void moe_cvt(const float* __restrict__ x, bf16* __restrict__ xh) {
    int i = (blockIdx.x * 256 + threadIdx.x) * 4;
    float4 v = *(const float4*)(x + i);
    v4bf o;
    o[0] = (bf16)v.x; o[1] = (bf16)v.y; o[2] = (bf16)v.z; o[3] = (bf16)v.w;
    *(v4bf*)(xh + i) = o;
}

// ---- Gate/Up GEMM + SiLU fuse: h[t,f] = silu(x.Wg)*(x.Wu), one expert per launch ----
// CTA tile M=128 x N=64, K-steps of 32 over D. Double-buffered LDS: async A-DMA and B global
// loads of step k+1 overlap the 8-WMMA burst of step k; one barrier per step.
__global__ __launch_bounds__(256) void moe_gu(const bf16* __restrict__ xh,
                                              const float* __restrict__ Wg,
                                              const float* __restrict__ Wu,
                                              bf16* __restrict__ hbuf, int e) {
    __shared__ bf16 la[2][128 * 40];
    __shared__ bf16 lbg[2][64 * 40];
    __shared__ bf16 lbu[2][64 * 40];
    int tid = threadIdx.x;
    int tBase = blockIdx.x * 128;
    int fBase = blockIdx.y * 64;
    int lane = tid & 31, r = lane & 15, half = lane >> 4;
    int m0 = (tid >> 5) * 16;
    v8f accG[4] = {};
    v8f accU[4] = {};
    const float* wgp = Wg + (size_t)e * D_DIM * F_DIM;
    const float* wup = Wu + (size_t)e * D_DIM * F_DIM;

    auto fillA = [&](int b, int d0) {
        int row0 = tid >> 2, kp0 = (tid & 3) * 8;
        int c1 = tid + 256;
        int row1 = c1 >> 2, kp1 = (c1 & 3) * 8;
        async_copy_b128(&la[b][row0 * 40 + kp0],
                        xh + (size_t)(tBase + row0) * D_DIM + d0 + kp0);
        async_copy_b128(&la[b][row1 * 40 + kp1],
                        xh + (size_t)(tBase + row1) * D_DIM + d0 + kp1);
    };
    auto loadB = [&](int d0, float2* gv, float2* uv) {
#pragma unroll
        for (int it = 0; it < 4; ++it) {
            int q = tid + it * 256;
            int row = q >> 5, c2 = q & 31;
            int f = fBase + c2 * 2;
            float2 zz; zz.x = 0.f; zz.y = 0.f;
            gv[it] = zz; uv[it] = zz;
            if (f < F_DIM) {   // F even, f even -> f+1 < F too
                gv[it] = *(const float2*)(wgp + (size_t)(d0 + row) * F_DIM + f);
                uv[it] = *(const float2*)(wup + (size_t)(d0 + row) * F_DIM + f);
            }
        }
    };
    auto storeB = [&](int b, const float2* gv, const float2* uv) {
#pragma unroll
        for (int it = 0; it < 4; ++it) {
            int q = tid + it * 256;
            int row = q >> 5, c2 = q & 31;
            lbg[b][(c2 * 2 + 0) * 40 + row] = (bf16)gv[it].x;
            lbg[b][(c2 * 2 + 1) * 40 + row] = (bf16)gv[it].y;
            lbu[b][(c2 * 2 + 0) * 40 + row] = (bf16)uv[it].x;
            lbu[b][(c2 * 2 + 1) * 40 + row] = (bf16)uv[it].y;
        }
    };

    {   // prologue: stage step 0 into buffer 0
        float2 gv[4], uv[4];
        fillA(0, 0);
        loadB(0, gv, uv);
        storeB(0, gv, uv);
        async_wait0();
        __syncthreads();
    }
    const int NSTEP = D_DIM / 32;
    for (int step = 0; step < NSTEP; ++step) {
        int cur = step & 1, nxt = cur ^ 1;
        bool hn = (step + 1) < NSTEP;
        float2 gv[4], uv[4];
        if (hn) {
            fillA(nxt, (step + 1) * 32);       // async DMA, lands during compute
            loadB((step + 1) * 32, gv, uv);    // global loads in flight during compute
        }
        v16bf a = frag_a(la[cur], m0 + r, 40, half);
        v16bf bg[4], bu[4];
#pragma unroll
        for (int nt = 0; nt < 4; nt++) {
            bg[nt] = frag_b(lbg[cur], nt * 16 + r, 40, half);
            bu[nt] = frag_b(lbu[cur], nt * 16 + r, 40, half);
        }
        __builtin_amdgcn_sched_barrier(0);     // group ds_loads; one dscnt wait for 8 WMMAs
#pragma unroll
        for (int nt = 0; nt < 4; nt++) {
            accG[nt] = __builtin_amdgcn_wmma_f32_16x16x32_bf16(false, a, false, bg[nt],
                                                               (short)0, accG[nt], false, false);
            accU[nt] = __builtin_amdgcn_wmma_f32_16x16x32_bf16(false, a, false, bu[nt],
                                                               (short)0, accU[nt], false, false);
        }
        if (hn) storeB(nxt, gv, uv);           // convert + LDS store for next step
        async_wait0();
        __syncthreads();
    }
    // epilogue: h = silu(g)*u, bf16 store
#pragma unroll
    for (int nt = 0; nt < 4; nt++) {
        int f = fBase + nt * 16 + r;
        if (f >= F_DIM) continue;
#pragma unroll
        for (int i = 0; i < 8; i++) {
            float g = accG[nt][i];
            float u = accU[nt][i];
            float h = (g / (1.0f + __expf(-g))) * u;
            int row = tBase + m0 + half * 8 + i;
            hbuf[(size_t)row * FP_DIM + f] = (bf16)h;
        }
    }
}

// ---- Down GEMM with combine: out[t,d] (+)= comb[t,e] * sum_f h[t,f]*Wd[e,f,d] ----
// CTA tile M=128 x N=128, K-steps of 32 over F (last step partial), double-buffered.
__global__ __launch_bounds__(256) void moe_down(const bf16* __restrict__ hbuf,
                                                const float* __restrict__ Wd,
                                                const float* __restrict__ comb,
                                                float* __restrict__ out, int e) {
    __shared__ bf16 la[2][128 * 40];
    __shared__ bf16 lbt[2][128 * 40];
    int tid = threadIdx.x;
    int tBase = blockIdx.x * 128;
    int dBase = blockIdx.y * 128;
    int lane = tid & 31, r = lane & 15, half = lane >> 4;
    int m0 = (tid >> 5) * 16;
    v8f acc[8] = {};
    const float* wdp = Wd + (size_t)e * F_DIM * D_DIM;

    auto fillA = [&](int b, int f0) {
        if (f0 + 32 <= F_DIM) {   // full chunk: async DMA
            int row0 = tid >> 2, kp0 = (tid & 3) * 8;
            int c1 = tid + 256;
            int row1 = c1 >> 2, kp1 = (c1 & 3) * 8;
            async_copy_b128(&la[b][row0 * 40 + kp0],
                            hbuf + (size_t)(tBase + row0) * FP_DIM + f0 + kp0);
            async_copy_b128(&la[b][row1 * 40 + kp1],
                            hbuf + (size_t)(tBase + row1) * FP_DIM + f0 + kp1);
        } else {                  // last partial chunk: guarded scalar path
            for (int c = tid; c < 512; c += 256) {
                int row = c >> 2, kp = (c & 3) * 8;
#pragma unroll
                for (int j = 0; j < 8; j++) {
                    int f = f0 + kp + j;
                    la[b][row * 40 + kp + j] =
                        (f < F_DIM) ? hbuf[(size_t)(tBase + row) * FP_DIM + f] : (bf16)0.0f;
                }
            }
        }
    };
    auto loadB = [&](int f0, float4* bv) {
#pragma unroll
        for (int it = 0; it < 4; ++it) {
            int q = tid + it * 256;
            int row = q >> 5, c4 = q & 31;
            int f = f0 + row;
            float4 zz; zz.x = zz.y = zz.z = zz.w = 0.f;
            bv[it] = zz;
            if (f < F_DIM) bv[it] = *(const float4*)(wdp + (size_t)f * D_DIM + dBase + c4 * 4);
        }
    };
    auto storeB = [&](int b, const float4* bv) {
#pragma unroll
        for (int it = 0; it < 4; ++it) {
            int q = tid + it * 256;
            int row = q >> 5, c4 = q & 31;
            lbt[b][(c4 * 4 + 0) * 40 + row] = (bf16)bv[it].x;
            lbt[b][(c4 * 4 + 1) * 40 + row] = (bf16)bv[it].y;
            lbt[b][(c4 * 4 + 2) * 40 + row] = (bf16)bv[it].z;
            lbt[b][(c4 * 4 + 3) * 40 + row] = (bf16)bv[it].w;
        }
    };

    const int NSTEP = (F_DIM + 31) / 32;   // 54
    {   // prologue: stage step 0
        float4 bv[4];
        fillA(0, 0);
        loadB(0, bv);
        storeB(0, bv);
        async_wait0();
        __syncthreads();
    }
    for (int step = 0; step < NSTEP; ++step) {
        int cur = step & 1, nxt = cur ^ 1;
        bool hn = (step + 1) < NSTEP;
        float4 bv[4];
        if (hn) {
            fillA(nxt, (step + 1) * 32);
            loadB((step + 1) * 32, bv);
        }
        v16bf a = frag_a(la[cur], m0 + r, 40, half);
        v16bf b[8];
#pragma unroll
        for (int nt = 0; nt < 8; nt++)
            b[nt] = frag_b(lbt[cur], nt * 16 + r, 40, half);
        __builtin_amdgcn_sched_barrier(0);
#pragma unroll
        for (int nt = 0; nt < 8; nt++)
            acc[nt] = __builtin_amdgcn_wmma_f32_16x16x32_bf16(false, a, false, b[nt],
                                                              (short)0, acc[nt], false, false);
        if (hn) storeB(nxt, bv);
        async_wait0();
        __syncthreads();
    }
    // epilogue: weight by comb and accumulate into out (e==0 initializes)
    float cw[8];
#pragma unroll
    for (int i = 0; i < 8; i++)
        cw[i] = comb[(size_t)(tBase + m0 + half * 8 + i) * E_EXP + e];
#pragma unroll
    for (int nt = 0; nt < 8; nt++) {
        int d = dBase + nt * 16 + r;
#pragma unroll
        for (int i = 0; i < 8; i++) {
            int row = tBase + m0 + half * 8 + i;
            size_t idx = (size_t)row * D_DIM + d;
            float v = cw[i] * acc[nt][i];
            if (e > 0) v += out[idx];
            out[idx] = v;
        }
    }
}

// ---- Finalize aux loss + mean probs ----
__global__ void moe_fin(const float* __restrict__ partials, float* __restrict__ tail) {
    if (threadIdx.x == 0) {
        float z = 0.0f, ps[E_EXP];
#pragma unroll
        for (int e = 0; e < E_EXP; e++) ps[e] = 0.0f;
        for (int b = 0; b < 16; b++) {
            z += partials[b * 9];
            for (int e = 0; e < E_EXP; e++) ps[e] += partials[b * 9 + 1 + e];
        }
        float lb = 0.0f;
        for (int e = 0; e < E_EXP; e++) {
            float mp = ps[e] / (float)T_TOK;
            tail[1 + e] = mp;
            lb += mp * mp;
        }
        float z_loss = z / (float)(T_TOK * E_EXP);
        tail[0] = (float)E_EXP * lb + 1e-3f * z_loss;
    }
}

extern "C" void kernel_launch(void* const* d_in, const int* in_sizes, int n_in,
                              void* d_out, int out_size, void* d_ws, size_t ws_size,
                              hipStream_t stream) {
    const float* x  = (const float*)d_in[0];
    const float* Wr = (const float*)d_in[1];
    const float* Wg = (const float*)d_in[2];
    const float* Wu = (const float*)d_in[3];
    const float* Wd = (const float*)d_in[4];
    float* out = (float*)d_out;

    char* ws = (char*)d_ws;
    float* comb     = (float*)ws;                          // T*E*4      = 128 KB
    float* partials = (float*)(ws + 131072);               // 16*9*4
    bf16*  xh       = (bf16*)(ws + 135168);                // T*D*2      = 8 MB
    bf16*  hbuf     = (bf16*)(ws + 135168 + 8388608);      // T*FP*2     = 14.2 MB

    moe_router<<<16, 256, 0, stream>>>(x, Wr, comb, partials);
    moe_cvt<<<(T_TOK * D_DIM) / (256 * 4), 256, 0, stream>>>(x, xh);
    moe_fin<<<1, 32, 0, stream>>>(partials, out + (size_t)T_TOK * D_DIM);

    for (int e = 0; e < E_EXP; e++) {
        moe_gu<<<dim3(T_TOK / 128, FP_DIM / 64), 256, 0, stream>>>(xh, Wg, Wu, hbuf, e);
        moe_down<<<dim3(T_TOK / 128, D_DIM / 128), 256, 0, stream>>>(hbuf, Wd, comb, out, e);
    }
}
